// PointWiseStereo_v3_26800595927730
// MI455X (gfx1250) — matrix operand order repaired and checked by reading, not compiled, not run on hardware
//
#include <hip/hip_runtime.h>
#include <math.h>

// ---------------- types for WMMA ----------------
typedef __attribute__((ext_vector_type(16))) __bf16 v16bf;
typedef __attribute__((ext_vector_type(8)))  float  v8f;
typedef __attribute__((ext_vector_type(2)))  float  v2f;

// ---------------- problem constants ----------------
#define H1   384
#define W1   1248
#define HW1  (H1*W1)          // 479232
#define H4   96
#define W4   312
#define HW4  (H4*W4)          // 29952
#define NPTS 2048
#define MAXD 192
#define BN_S 0.999995000037f  // 1/sqrt(1+1e-5)

// ================= resize (quarter scale, align_corners bilinear) =================
__global__ void resize4_kernel(const float* __restrict__ L, const float* __restrict__ R,
                               float* __restrict__ oL, float* __restrict__ oR) {
  int idx = blockIdx.x * blockDim.x + threadIdx.x;
  int total = 2 * 3 * HW4;
  if (idx >= total) return;
  int im = idx / (3 * HW4);
  int r  = idx - im * (3 * HW4);
  int c  = r / HW4;
  int p  = r - c * HW4;
  int oy = p / W4, ox = p - (p / W4) * W4;
  float ys = (float)oy * (float)(H1 - 1) / (float)(H4 - 1);
  float xs = (float)ox * (float)(W1 - 1) / (float)(W4 - 1);
  int y0 = (int)floorf(ys); int y1 = min(y0 + 1, H1 - 1);
  int x0 = (int)floorf(xs); int x1 = min(x0 + 1, W1 - 1);
  float fy = ys - (float)y0, fx = xs - (float)x0;
  const float* s = (im ? R : L) + c * HW1;
  float top = s[y0 * W1 + x0] * (1.f - fx) + s[y0 * W1 + x1] * fx;
  float bot = s[y1 * W1 + x0] * (1.f - fx) + s[y1 * W1 + x1] * fx;
  (im ? oR : oL)[c * HW4 + p] = top * (1.f - fy) + bot * fy;
}

// ================= fused conv7x7 (3->32) + conv1x1 (32->32), WMMA bf16 =================
// One wave per 16-pixel M-tile; both 16-channel N-halves per wave.
// conv7 K = 3*49 = 147 padded to 160 (5 steps of 32); D-tile is re-fed through a
// wave-local LDS transpose tile as conv1's A fragment (no HBM tmp round-trip).
template<int H, int W>
__global__ void fen_fused_wmma_kernel(const float* __restrict__ img,
                                      const float* __restrict__ w7, // [32][147]
                                      const float* __restrict__ w1, // [32][32]
                                      __bf16* __restrict__ feat) {  // [32][HW]
  constexpr int HW = H * W;
  constexpr int Mtiles = (HW + 15) / 16;
  __shared__ __bf16 ldsT[8][32][16];   // per-wave 32ch x 16px staging (8 KB)

  const int lane = threadIdx.x & 31;
  const int wv   = threadIdx.x >> 5;
  const int gw = (blockIdx.x * blockDim.x + threadIdx.x) >> 5;
  if (gw >= Mtiles) return;
  const int mbase = gw << 4;
  const int half = lane >> 4;
  const int col = lane & 15;
  const int y0 = mbase / W;
  const int x0 = mbase - y0 * W;
  const bool interior = (x0 >= 3) && (x0 + 18 < W) && (y0 >= 3) && (y0 + 4 <= H) &&
                        (mbase + 15 < HW);
  const float* wb0 = w7 + col * 147;
  const float* wb1 = w7 + (col + 16) * 147;

  v8f acc0 = {}, acc1 = {};
  if (interior) {
    // fast path: unconditional loads, compile-time offsets (one cndmask per element)
    const float* base = img + (y0 - 3) * W + (x0 + col - 3);
#pragma unroll
    for (int kb = 0; kb < 5; ++kb) {
      v16bf a, b0, b1;
#pragma unroll
      for (int e = 0; e < 16; ++e) {
        const int ka0 = kb * 32 + (e & 7) + ((e >> 3) << 4);
        const int ka1 = ka0 + 8;
        const int off0 = (ka0 < 147) ? ((ka0 / 49) * HW + ((ka0 % 49) / 7) * W + (ka0 % 7)) : 0;
        const int off1 = (ka1 < 147) ? ((ka1 / 49) * HW + ((ka1 % 49) / 7) * W + (ka1 % 7)) : 0;
        a[e] = (__bf16)base[half ? off1 : off0];   // k>=147 pairs with B==0
        const int kB0 = kb * 32 + (e & 7) + ((e >> 3) << 4), kB1 = kB0 + 8;
        const int kB = half ? kB1 : kB0;
        const bool bval = half ? (kB1 < 147) : (kB0 < 147);
        b0[e] = (__bf16)(bval ? wb0[kB] : 0.f);
        b1[e] = (__bf16)(bval ? wb1[kB] : 0.f);
      }
      acc0 = __builtin_amdgcn_wmma_f32_16x16x32_bf16(false, a, false, b0, (short)0, acc0, false, false);
      acc1 = __builtin_amdgcn_wmma_f32_16x16x32_bf16(false, a, false, b1, (short)0, acc1, false, false);
    }
  } else {
    // border path: select between two fully-formed clamped offsets + validity bits
    const int m = mbase + col;
    const int mc = m < HW ? m : HW - 1;
    const int yy = mc / W;
    const int xx = mc - yy * W;
    const bool mval = m < HW;
#pragma unroll
    for (int kb = 0; kb < 5; ++kb) {
      v16bf a, b0, b1;
#pragma unroll
      for (int e = 0; e < 16; ++e) {
        const int ka0 = kb * 32 + (e & 7) + ((e >> 3) << 4);
        const int ka1 = ka0 + 8;
        const int c0 = ka0 / 49, kh0 = (ka0 % 49) / 7, kw0 = ka0 % 7;
        const int c1 = ka1 / 49, kh1 = (ka1 % 49) / 7, kw1 = ka1 % 7;
        const int iy0 = yy + kh0 - 3, ix0 = xx + kw0 - 3;
        const int iy1 = yy + kh1 - 3, ix1 = xx + kw1 - 3;
        const bool v0 = (ka0 < 147) && iy0 >= 0 && iy0 < H && ix0 >= 0 && ix0 < W;
        const bool v1 = (ka1 < 147) && iy1 >= 0 && iy1 < H && ix1 >= 0 && ix1 < W;
        const int o0 = v0 ? (c0 * HW + iy0 * W + ix0) : 0;
        const int o1 = v1 ? (c1 * HW + iy1 * W + ix1) : 0;
        const bool vv = (half ? v1 : v0) && mval;
        const float av = vv ? img[half ? o1 : o0] : 0.f;
        a[e] = (__bf16)av;
        const int kB0 = kb * 32 + (e & 7) + ((e >> 3) << 4), kB1 = kB0 + 8;
        const int kB = half ? kB1 : kB0;
        const bool bval = half ? (kB1 < 147) : (kB0 < 147);
        b0[e] = (__bf16)(bval ? wb0[kB] : 0.f);
        b1[e] = (__bf16)(bval ? wb1[kB] : 0.f);
      }
      acc0 = __builtin_amdgcn_wmma_f32_16x16x32_bf16(false, a, false, b0, (short)0, acc0, false, false);
      acc1 = __builtin_amdgcn_wmma_f32_16x16x32_bf16(false, a, false, b1, (short)0, acc1, false, false);
    }
  }

  // relu(BN_S * conv7) -> wave-local LDS transpose tile (pixels x channels -> A layout)
#pragma unroll
  for (int r = 0; r < 8; ++r) {
    const int p = r + (half << 3);      // pixel index within tile (D row)
    float v0 = acc0[r] * BN_S; v0 = v0 > 0.f ? v0 : 0.f;
    float v1 = acc1[r] * BN_S; v1 = v1 > 0.f ? v1 : 0.f;
    ldsT[wv][col][p]      = (__bf16)v0;
    ldsT[wv][col + 16][p] = (__bf16)v1;
  }
  // conv1x1: A[m=col][k=channel], single K=32 step, both N-halves
  v16bf a1, c0v, c1v;
#pragma unroll
  for (int e = 0; e < 16; ++e) {
    const int k0 = (e & 7) + ((e >> 3) << 4);
    const int kk = half ? (k0 + 8) : k0;
    a1[e] = ldsT[wv][kk][col];
    const int kv = half ? (k0 + 8) : k0;
    c0v[e] = (__bf16)w1[col * 32 + kv];
    c1v[e] = (__bf16)w1[(col + 16) * 32 + kv];
  }
  v8f f0 = {}, f1 = {};
  f0 = __builtin_amdgcn_wmma_f32_16x16x32_bf16(false, a1, false, c0v, (short)0, f0, false, false);
  f1 = __builtin_amdgcn_wmma_f32_16x16x32_bf16(false, a1, false, c1v, (short)0, f1, false, false);
#pragma unroll
  for (int r = 0; r < 8; ++r) {
    const int mm = mbase + r + (half << 3);
    if (mm < HW) {
      feat[col * HW + mm]        = (__bf16)f0[r];
      feat[(col + 16) * HW + mm] = (__bf16)f1[r];
    }
  }
}

// ================= bilinear sampler (zero-weight outside bounds) =================
struct Samp { int i0, i1, i2, i3; float w0, w1, w2, w3; };

__device__ __forceinline__ Samp mk_samp(float sxp, float syp, int Hf, int Wf) {
  float x0f = floorf(sxp), y0f = floorf(syp);
  int x0 = (int)x0f, y0 = (int)y0f;
  float wx = sxp - x0f, wy = syp - y0f;
  int   xs[2]  = {x0, x0 + 1}, ys[2] = {y0, y0 + 1};
  float wxs[2] = {1.f - wx, wx}, wys[2] = {1.f - wy, wy};
  int idx[4]; float w[4]; int q = 0;
#pragma unroll
  for (int dy = 0; dy < 2; ++dy)
#pragma unroll
    for (int dx = 0; dx < 2; ++dx) {
      int xi = xs[dx], yi = ys[dy];
      bool valid = (xi >= 0) && (xi < Wf) && (yi >= 0) && (yi < Hf);
      int xc = min(max(xi, 0), Wf - 1);
      int yc = min(max(yi, 0), Hf - 1);
      idx[q] = yc * Wf + xc;
      w[q]   = valid ? wxs[dx] * wys[dy] : 0.f;
      ++q;
    }
  Samp s; s.i0 = idx[0]; s.i1 = idx[1]; s.i2 = idx[2]; s.i3 = idx[3];
  s.w0 = w[0]; s.w1 = w[1]; s.w2 = w[2]; s.w3 = w[3];
  return s;
}

__device__ __forceinline__ float ap_samp(const Samp& s, const __bf16* f) {
  return (float)f[s.i0] * s.w0 + (float)f[s.i1] * s.w1 +
         (float)f[s.i2] * s.w2 + (float)f[s.i3] * s.w3;
}

// ================= block matching: 1 workgroup per point =================
// cost layout: [n][8 groups][192 disparities]
__global__ void block_match_kernel(const __bf16* __restrict__ fL1, const __bf16* __restrict__ fR1,
                                   const __bf16* __restrict__ fL4, const __bf16* __restrict__ fR4,
                                   const float* __restrict__ points,
                                   float* __restrict__ cost) {
  __shared__ float lv1[25][32];
  __shared__ float lv4[25][32];
  __shared__ float c4[8][48];

  int n = blockIdx.x;
  int t = threadIdx.x;
  float px = points[n * 2 + 0], py = points[n * 2 + 1];
  const float sx4 = (float)(W4 - 1) / (float)(W1 - 1);
  const float sy4 = (float)(H4 - 1) / (float)(H1 - 1);

  if (t < 25) {
    int w = t;
    float ox = (float)((w % 5) - 2), oy = (float)((w / 5) - 2);
    Samp s = mk_samp(px + ox, py + oy, H1, W1);
    for (int ch = 0; ch < 32; ++ch) lv1[w][ch] = ap_samp(s, fL1 + ch * HW1);
  } else if (t < 50) {
    int w = t - 25;
    float ox = (float)(((w % 5) - 2) * 4), oy = (float)(((w / 5) - 2) * 4);
    Samp s = mk_samp((px + ox) * sx4, (py + oy) * sy4, H4, W4);
    for (int ch = 0; ch < 32; ++ch) lv4[w][ch] = ap_samp(s, fL4 + ch * HW4);
  }
  __syncthreads();

  float cst1[8];
  if (t < 192) {
    float acc[8];
#pragma unroll
    for (int g = 0; g < 8; ++g) acc[g] = 0.f;
    float d = (float)t;
    for (int w = 0; w < 25; ++w) {
      float ox = (float)((w % 5) - 2), oy = (float)((w / 5) - 2);
      Samp s = mk_samp(px + ox - d, py + oy, H1, W1);
      __builtin_prefetch(fR1 + s.i0, 0, 0);
#pragma unroll
      for (int g = 0; g < 8; ++g)
#pragma unroll
        for (int c = 0; c < 4; ++c) {
          int ch = g * 4 + c;
          float rv = ap_samp(s, fR1 + ch * HW1);
          acc[g] += fabsf(lv1[w][ch] - rv);
        }
    }
#pragma unroll
    for (int g = 0; g < 8; ++g) cst1[g] = 1.f - expf(-acc[g] * 0.01f);
  }

  if (t < 48) {
    float acc[8];
#pragma unroll
    for (int g = 0; g < 8; ++g) acc[g] = 0.f;
    float d = (float)(t * 4);
    for (int w = 0; w < 25; ++w) {
      float ox = (float)(((w % 5) - 2) * 4), oy = (float)(((w / 5) - 2) * 4);
      Samp s = mk_samp((px + ox - d) * sx4, (py + oy) * sy4, H4, W4);
#pragma unroll
      for (int g = 0; g < 8; ++g)
#pragma unroll
        for (int c = 0; c < 4; ++c) {
          int ch = g * 4 + c;
          float rv = ap_samp(s, fR4 + ch * HW4);
          acc[g] += fabsf(lv4[w][ch] - rv);
        }
    }
#pragma unroll
    for (int g = 0; g < 8; ++g) c4[g][t] = 1.f - expf(-acc[g] * 0.01f);
  }
  __syncthreads();

  if (t < 192) {
    float xs = (float)t * 47.f / 191.f;
    float x0f = floorf(xs);
    int x0 = (int)x0f;
    int x1 = min(x0 + 1, 47);
    float f = xs - x0f;
#pragma unroll
    for (int g = 0; g < 8; ++g) {
      float v4 = c4[g][x0] * (1.f - f) + c4[g][x1] * f;
      cost[(n * 8 + g) * MAXD + t] = cst1[g] + v4;
    }
  }
}

// ================= aggregation on the matrix engine =================
// Each conv-along-d layer as GEMM: M=16-disparity tile, N=8 out channels (pad 16),
// K=24 (8 in-ch x 3 taps) -> 6x V_WMMA_F32_16X16X4_F32 (fp32, no precision loss).
// LDS arrays are halo-padded ([8][DST], d stored at d+1, halo zero) so every A load
// is one unconditional ds_load; weights padded to 16 oc so B loads are unconditional.
#define DST 200   // 192 + 2 halo, padded stride

__device__ __forceinline__ void agg_layer(const float* __restrict__ src,   // [8*DST]
                                          float* __restrict__ dst,         // [8*DST]
                                          const float* __restrict__ wshP,  // [8][16][24]
                                          int layer, int dbase, int lane, bool addres) {
  const int half = lane >> 4;
  const int col = lane & 15;
  const int dcol = dbase + col;          // A row's padded base index (d+1 handled below)
  const int wbase = layer * 384 + col * 24;
  v8f acc = {};
#pragma unroll
  for (int ks = 0; ks < 6; ++ks) {
    v2f A, B;
#pragma unroll
    for (int v = 0; v < 2; ++v) {
      const int k0 = ks * 4 + v;         // K for lanes 0-15
      const int k1 = k0 + 2;             // K for lanes 16-31
      // padded A offset: ic*DST + (d+1) with d = dbase+col+tp-1  ->  ic*DST + tp + dcol
      const int aoff0 = (k0 / 3) * DST + (k0 % 3);
      const int aoff1 = (k1 / 3) * DST + (k1 % 3);
      A[v] = src[dcol + (half ? aoff1 : aoff0)];
      B[v] = wshP[wbase + (half ? k1 : k0)];   // ic*3+tp == k: no division anywhere
    }
    acc = __builtin_amdgcn_wmma_f32_16x16x4_f32(false, A, false, B, (short)0, acc, false, false);
  }
#pragma unroll
  for (int r = 0; r < 8; ++r) {
    const int d = dbase + r + (half << 3);
    if (col < 8) {
      float s = acc[r] * BN_S;
      s = s > 0.f ? s : 0.f;
      if (addres) dst[col * DST + d + 1] += s;
      else        dst[col * DST + d + 1] = s;
    }
  }
}

__global__ void agg_topk_kernel(const float* __restrict__ cost,
                                const float* __restrict__ agg_w,   // [8][8][8][3]
                                const float* __restrict__ agg_w11, // [8]
                                float* __restrict__ out) {
  __shared__ float aP[8 * DST];
  __shared__ float tP[8 * DST];
  __shared__ float wshP[8 * 16 * 24];   // oc padded to 16 (zeros)
  __shared__ float fin[MAXD];

  const int n = blockIdx.x, tid = threadIdx.x;   // 384 threads = 12 wave32
  const int lane = tid & 31;
  const int dbase = (tid >> 5) << 4;             // 16 disparities per wave

  for (int i = tid; i < 8 * DST; i += 384) { aP[i] = 0.f; tP[i] = 0.f; }
  for (int i = tid; i < 8 * 16 * 24; i += 384) {
    int layer = i / 384, rem = i - layer * 384;
    int oc = rem / 24, k = rem - oc * 24;
    wshP[i] = (oc < 8) ? agg_w[layer * 192 + oc * 24 + k] : 0.f;
  }
  __syncthreads();
  for (int i = tid; i < 8 * MAXD; i += 384) {
    int ch = i / MAXD, d = i - ch * MAXD;
    aP[ch * DST + d + 1] = cost[n * (8 * MAXD) + i];
  }
  __syncthreads();

#pragma unroll
  for (int pair = 0; pair < 4; ++pair) {
    agg_layer(aP, tP, wshP, pair * 2, dbase, lane, false);      // x_{2k+1} = bc(x_{2k})
    __syncthreads();
    agg_layer(tP, aP, wshP, pair * 2 + 1, dbase, lane, true);   // x_{2k+2} = bc(.) + x_{2k}
    __syncthreads();
  }

  if (tid < MAXD) {
    float s = 0.f;
#pragma unroll
    for (int ic = 0; ic < 8; ++ic) s += agg_w11[ic] * aP[ic * DST + tid + 1];
    fin[tid] = s;
  }
  __syncthreads();

  if (tid == 0) {
    float v0 = -1e30f, v1 = -1e30f; int i0 = 0, i1 = 0;
    for (int i = 0; i < MAXD; ++i) {
      float v = fin[i];
      if (v > v0)      { v1 = v0; i1 = i0; v0 = v; i0 = i; }
      else if (v > v1) { v1 = v;  i1 = i; }
    }
    float p0 = 1.f / (1.f + expf(v1 - v0));
    out[n] = (float)i0 * p0 + (float)i1 * (1.f - p0);
  }
}

// ================= host launch =================
extern "C" void kernel_launch(void* const* d_in, const int* in_sizes, int n_in,
                              void* d_out, int out_size, void* d_ws, size_t ws_size,
                              hipStream_t stream) {
  (void)in_sizes; (void)n_in; (void)out_size; (void)ws_size;
  const float* left    = (const float*)d_in[0];
  const float* right   = (const float*)d_in[1];
  const float* points  = (const float*)d_in[2];
  const float* fen_w7  = (const float*)d_in[3]; // [2][32][147]
  const float* fen_w1  = (const float*)d_in[4]; // [2][32][32]
  const float* agg_w   = (const float*)d_in[5]; // [8][8][8][3]
  const float* agg_w11 = (const float*)d_in[6]; // [8]
  float* out = (float*)d_out;

  char* ws = (char*)d_ws;
  size_t off = 0;
  auto carve = [&](size_t bytes) -> void* {
    void* p = ws + off;
    off = (off + bytes + 255) & ~(size_t)255;
    return p;
  };
  float*  img4L  = (float*) carve((size_t)3 * HW4 * sizeof(float));
  float*  img4R  = (float*) carve((size_t)3 * HW4 * sizeof(float));
  __bf16* featL1 = (__bf16*)carve((size_t)32 * HW1 * sizeof(__bf16));
  __bf16* featR1 = (__bf16*)carve((size_t)32 * HW1 * sizeof(__bf16));
  __bf16* featL4 = (__bf16*)carve((size_t)32 * HW4 * sizeof(__bf16));
  __bf16* featR4 = (__bf16*)carve((size_t)32 * HW4 * sizeof(__bf16));
  float*  cost   = (float*) carve((size_t)NPTS * 8 * MAXD * sizeof(float));

  // 1) quarter-scale images
  {
    int tot = 2 * 3 * HW4;
    resize4_kernel<<<(tot + 255) / 256, 256, 0, stream>>>(left, right, img4L, img4R);
  }

  // 2) fused feature extraction (conv7 WMMA -> LDS transpose -> conv1 WMMA)
  {
    constexpr int Mt1 = (HW1 + 15) / 16;
    constexpr int Mt4 = (HW4 + 15) / 16;
    const int b1 = (Mt1 + 7) / 8;   // 8 wave32 per 256-thread block
    const int b4 = (Mt4 + 7) / 8;
    fen_fused_wmma_kernel<H1, W1><<<b1, 256, 0, stream>>>(left,  fen_w7, fen_w1, featL1);
    fen_fused_wmma_kernel<H1, W1><<<b1, 256, 0, stream>>>(right, fen_w7, fen_w1, featR1);
    fen_fused_wmma_kernel<H4, W4><<<b4, 256, 0, stream>>>(img4L, fen_w7 + 32 * 147, fen_w1 + 32 * 32, featL4);
    fen_fused_wmma_kernel<H4, W4><<<b4, 256, 0, stream>>>(img4R, fen_w7 + 32 * 147, fen_w1 + 32 * 32, featR4);
  }

  // 3) block matching (one workgroup per point; bf16 feats are L2-resident)
  block_match_kernel<<<NPTS, 256, 0, stream>>>(featL1, featR1, featL4, featR4, points, cost);

  // 4) aggregation (fp32 WMMA) + top-2 softmax blend
  agg_topk_kernel<<<NPTS, 384, 0, stream>>>(cost, agg_w, agg_w11, out);
}
